// MultiHeadAttention_66116726555211
// MI455X (gfx1250) — matrix operand order repaired and checked by reading, not compiled, not run on hardware
//
#include <hip/hip_runtime.h>
#include <hip/hip_bf16.h>

typedef __attribute__((ext_vector_type(16))) __bf16 bf16x16;
typedef __attribute__((ext_vector_type(8)))  __bf16 bf16x8;
typedef __attribute__((ext_vector_type(4)))  __bf16 bf16x4;
typedef __attribute__((ext_vector_type(8)))  float  f32x8;

union AB16 { bf16x16 v; bf16x8 h[2]; };

__device__ __forceinline__ f32x8 wmma_bf16(const AB16& a, const AB16& b, f32x8 c) {
  return __builtin_amdgcn_wmma_f32_16x16x32_bf16(false, a.v, false, b.v,
                                                 (short)0, c, false, false);
}

__device__ __forceinline__ f32x8 f32x8_zero() {
  f32x8 z = {0.f, 0.f, 0.f, 0.f, 0.f, 0.f, 0.f, 0.f};
  return z;
}

// LDS byte offset of a generic pointer to __shared__ (ISA 10.2: LDS generic
// address carries the wave-relative LDS byte address in addr[31:0]).
__device__ __forceinline__ unsigned lds_off(const void* p) {
  return (unsigned)(unsigned long long)p;
}

// Async global -> LDS copy, 16B per lane, tracked by ASYNCcnt.
__device__ __forceinline__ void async_ld_b128(unsigned lds, const void* g) {
  unsigned long long ga = (unsigned long long)g;
  asm volatile("global_load_async_to_lds_b128 %0, %1, off"
               :: "v"(lds), "v"(ga) : "memory");
}
__device__ __forceinline__ void wait_async0() {
  asm volatile("s_wait_asynccnt 0x0" ::: "memory");
}
__device__ __forceinline__ void wait_ds0() {
  asm volatile("s_wait_dscnt 0x0" ::: "memory");
}

// ---------------------------------------------------------------------------
// f32 -> bf16 conversion (linear; n multiple of 1024)
// ---------------------------------------------------------------------------
__global__ __launch_bounds__(256) void cvt_f32_to_bf16(const float* __restrict__ in,
                                                       __bf16* __restrict__ out) {
  int i = (blockIdx.x * 256 + threadIdx.x) * 4;
  float4 f = *(const float4*)(in + i);
  bf16x4 o;
  o.x = (__bf16)f.x; o.y = (__bf16)f.y; o.z = (__bf16)f.z; o.w = (__bf16)f.w;
  *(bf16x4*)(out + i) = o;
}

// f32 [K][N] -> bf16 transposed [N][K] (one-time weight prep)
__global__ __launch_bounds__(256)
void cvt_transpose_bf16(const float* __restrict__ in, __bf16* __restrict__ out,
                        int N, int K) {
  int n = blockIdx.x * 256 + threadIdx.x;
  int k = blockIdx.y;
  out[(size_t)n * K + k] = (__bf16)in[(size_t)k * N + n];
}

// ---------------------------------------------------------------------------
// Tiled WMMA GEMM:  C[M,N] = A[M,K] * Bt[N,K]^T + bias[N]
// 256 threads = 8 waves; WG tile 128(M) x 128(N); wave tile 32x64; K-step 32.
// Both tiles staged via async global->LDS, double-buffered.
// SPLIT_V: columns n >= 2048 are the V third of the QKV projection and are
// written transposed into vT[b][head][d][token] for the attention kernel.
// ---------------------------------------------------------------------------
template <typename OutT, bool SPLIT_V>
__global__ __launch_bounds__(256)
void gemm_bf16_wmma(const __bf16* __restrict__ A, const __bf16* __restrict__ Bt,
                    const float* __restrict__ bias, OutT* __restrict__ C,
                    __bf16* __restrict__ vT,
                    int K, int lda, int ldbt, int ldc) {
  __shared__ __bf16 ldsA[2][128][40];   // [buf][m][k], row stride 80B
  __shared__ __bf16 ldsBT[2][128][40];  // [buf][n][k]

  const int tid   = threadIdx.x;
  const int lane  = tid & 31;
  const int wave  = tid >> 5;
  const int lm    = lane & 15;
  const int h8    = lane >> 4;
  const int waveM = wave & 3;        // 4 waves along M, 32 rows each
  const int waveN = wave >> 2;       // 2 waves along N, 64 cols each
  const int m0    = blockIdx.y * 128;
  const int n0    = blockIdx.x * 128;

  f32x8 acc[2][4];
  #pragma unroll
  for (int i = 0; i < 2; ++i)
    #pragma unroll
    for (int j = 0; j < 4; ++j) acc[i][j] = f32x8_zero();

  auto issueTiles = [&](int k0, int buf) {
    #pragma unroll
    for (int c = tid; c < 512; c += 256) {       // 128x32 bf16 = 512 x 16B each
      int row = c >> 2;
      int kc  = (c & 3) * 8;
      async_ld_b128(lds_off(&ldsA[buf][row][kc]),
                    &A[(size_t)(m0 + row) * lda + k0 + kc]);
      async_ld_b128(lds_off(&ldsBT[buf][row][kc]),
                    &Bt[(size_t)(n0 + row) * ldbt + k0 + kc]);
    }
  };

  issueTiles(0, 0);

  const int KT = K >> 5;
  for (int kt = 0; kt < KT; ++kt) {
    const int cur = kt & 1;
    wait_async0();                    // tile kt landed
    __syncthreads();                  // visible to all; prior compute done
    if (kt + 1 < KT) issueTiles((kt + 1) * 32, cur ^ 1);

    AB16 af[2];
    #pragma unroll
    for (int i = 0; i < 2; ++i) {
      const __bf16* ar = &ldsA[cur][waveM * 32 + i * 16 + lm][0];
      af[i].h[0] = *(const bf16x8*)(ar + h8 * 8);
      af[i].h[1] = *(const bf16x8*)(ar + 16 + h8 * 8);
    }
    #pragma unroll
    for (int j = 0; j < 4; ++j) {
      AB16 bfr;
      const __bf16* br = &ldsBT[cur][waveN * 64 + j * 16 + lm][0];
      bfr.h[0] = *(const bf16x8*)(br + h8 * 8);
      bfr.h[1] = *(const bf16x8*)(br + 16 + h8 * 8);
      #pragma unroll
      for (int i = 0; i < 2; ++i)
        acc[i][j] = wmma_bf16(af[i], bfr, acc[i][j]);
    }
  }

  // Epilogue: lane holds col lm, VGPR r holds row r + h8*8
  #pragma unroll
  for (int j = 0; j < 4; ++j) {
    int n = n0 + waveN * 64 + j * 16 + lm;
    float bv = bias[n];
    #pragma unroll
    for (int i = 0; i < 2; ++i) {
      #pragma unroll
      for (int r = 0; r < 8; ++r) {
        int m = m0 + waveM * 32 + i * 16 + h8 * 8 + r;
        float v = acc[i][j][r] + bv;
        if constexpr (SPLIT_V) {
          if (n >= 2048) {  // V third -> transposed layout [b][head][d][token]
            int head = (n - 2048) >> 6;
            int d    = (n - 2048) & 63;
            int bb   = m >> 11;
            int t    = m & 2047;
            vT[(((size_t)bb * 16 + head) * 64 + d) * 2048 + t] = (__bf16)v;
            continue;
          }
        }
        C[(size_t)m * ldc + n] = (OutT)v;
      }
    }
  }
}

// ---------------------------------------------------------------------------
// Causal flash attention. qkv [4096][3072] bf16 holds Q (cols 0..1023) and
// K (cols 1024..2047); V comes pre-transposed from vT [2][16][64][2048].
// One WG = 64 query rows of one (b, head); 128 threads = 4 waves.
// Q/K/V all staged with async global->LDS; K/V double-buffered.
// ---------------------------------------------------------------------------
__global__ __launch_bounds__(128)
void flash_attn_wmma(const __bf16* __restrict__ qkv, const __bf16* __restrict__ vT,
                     __bf16* __restrict__ out) {
  __shared__ __bf16 Ks[2][64][72];     // [buf][key][d]; Ks[1] doubles as Q stage
  __shared__ __bf16 Vt[2][64][72];     // [buf][d][key]
  __shared__ float  Ss[4][16][64];
  __shared__ __bf16 Ps[4][16][72];
  __shared__ float  rowScale[4][16];
  __shared__ float  rowInv[4][16];

  const int tid  = threadIdx.x;
  const int lane = tid & 31;
  const int wave = tid >> 5;
  const int lm   = lane & 15;
  const int h8   = lane >> 4;
  const int qi   = blockIdx.x;
  const int head = blockIdx.y;
  const int b    = blockIdx.z;
  const int q0   = qi * 64;
  const size_t ldq = 3072;
  const __bf16* qbase = qkv + (size_t)(b * 2048 + q0) * ldq + head * 64;
  const __bf16* vbase = vT + (size_t)(b * 16 + head) * 64 * 2048;

  auto issueKV = [&](int kt, int buf) {
    const __bf16* kb = qkv + (size_t)(b * 2048 + kt * 64) * ldq + 1024 + head * 64;
    const __bf16* vb = vbase + kt * 64;
    #pragma unroll
    for (int c = tid; c < 512; c += 128) {
      int row = c >> 3;            // key row (K) / d row (V)
      int dc  = (c & 7) * 8;       // d chunk (K) / key chunk (V)
      async_ld_b128(lds_off(&Ks[buf][row][dc]), kb + (size_t)row * ldq + dc);
      async_ld_b128(lds_off(&Vt[buf][row][dc]), vb + (size_t)row * 2048 + dc);
    }
  };

  // Prologue: Q -> Ks[1] (released after fragments hoisted), KV(0) -> buf 0
  #pragma unroll
  for (int c = tid; c < 512; c += 128) {
    int row = c >> 3;
    int dc  = (c & 7) * 8;
    async_ld_b128(lds_off(&Ks[1][row][dc]), qbase + (size_t)row * ldq + dc);
  }
  issueKV(0, 0);
  wait_async0();
  __syncthreads();

  // Hoist Q fragments (loop-invariant A operand)
  AB16 aq[2];
  #pragma unroll
  for (int kk = 0; kk < 2; ++kk) {
    const __bf16* qr = &Ks[1][wave * 16 + lm][kk * 32];
    aq[kk].h[0] = *(const bf16x8*)(qr + h8 * 8);
    aq[kk].h[1] = *(const bf16x8*)(qr + 16 + h8 * 8);
  }
  wait_ds0();   // Q reads complete before Ks[1] may be overwritten

  f32x8 o[4];
  #pragma unroll
  for (int j = 0; j < 4; ++j) o[j] = f32x8_zero();
  float rM = -__builtin_inff();
  float rL = 0.0f;

  for (int kt = 0; kt <= qi; ++kt) {
    const int cur = kt & 1;
    wait_async0();                       // KV(kt) landed (no-op for kt==0)
    __syncthreads();                     // visible to all; prior reads done
    if (kt < qi) issueKV(kt + 1, cur ^ 1);

    // ---- S = Q * K^T ----
    #pragma unroll
    for (int j = 0; j < 4; ++j) {
      f32x8 s = f32x8_zero();
      #pragma unroll
      for (int kk = 0; kk < 2; ++kk) {
        AB16 bk;
        const __bf16* kr = &Ks[cur][j * 16 + lm][kk * 32];
        bk.h[0] = *(const bf16x8*)(kr + h8 * 8);
        bk.h[1] = *(const bf16x8*)(kr + 16 + h8 * 8);
        s = wmma_bf16(aq[kk], bk, s);
      }
      #pragma unroll
      for (int r = 0; r < 8; ++r)
        Ss[wave][h8 * 8 + r][j * 16 + lm] = s[r];
    }
    __syncthreads();

    // ---- online softmax: lane < 16 handles one query row ----
    if (lane < 16) {
      const int rloc  = lane;
      const int qglob = q0 + wave * 16 + rloc;
      float tmax = -__builtin_inff();
      for (int jj = 0; jj < 64; ++jj) {
        float x = Ss[wave][rloc][jj] * 0.125f;            // 1/sqrt(64)
        if (kt * 64 + jj > qglob) x = -__builtin_inff();  // causal mask
        tmax = fmaxf(tmax, x);
      }
      float nM    = fmaxf(rM, tmax);
      float alpha = __expf(rM - nM);
      float psum  = 0.0f;
      for (int jj = 0; jj < 64; ++jj) {
        float x = Ss[wave][rloc][jj] * 0.125f;
        if (kt * 64 + jj > qglob) x = -__builtin_inff();
        float p = __expf(x - nM);
        psum += p;
        Ps[wave][rloc][jj] = (__bf16)p;
      }
      rL = rL * alpha + psum;
      rM = nM;
      rowScale[wave][rloc] = alpha;
    }
    __syncthreads();

    // ---- rescale O and accumulate O += P * V ----
    #pragma unroll
    for (int j = 0; j < 4; ++j) {
      #pragma unroll
      for (int r = 0; r < 8; ++r)
        o[j][r] *= rowScale[wave][h8 * 8 + r];
    }
    AB16 ap[2];
    #pragma unroll
    for (int kk = 0; kk < 2; ++kk) {
      const __bf16* pr = &Ps[wave][lm][kk * 32];
      ap[kk].h[0] = *(const bf16x8*)(pr + h8 * 8);
      ap[kk].h[1] = *(const bf16x8*)(pr + 16 + h8 * 8);
    }
    #pragma unroll
    for (int j = 0; j < 4; ++j) {
      #pragma unroll
      for (int kk = 0; kk < 2; ++kk) {
        AB16 bv;
        const __bf16* vr = &Vt[cur][j * 16 + lm][kk * 32];
        bv.h[0] = *(const bf16x8*)(vr + h8 * 8);
        bv.h[1] = *(const bf16x8*)(vr + 16 + h8 * 8);
        o[j] = wmma_bf16(ap[kk], bv, o[j]);
      }
    }
  }

  // ---- finalize: O /= rowSum, store bf16 ----
  if (lane < 16) rowInv[wave][lane] = 1.0f / rL;
  __syncthreads();
  #pragma unroll
  for (int j = 0; j < 4; ++j) {
    #pragma unroll
    for (int r = 0; r < 8; ++r) {
      int row = q0 + wave * 16 + h8 * 8 + r;
      int col = head * 64 + j * 16 + lm;
      out[(size_t)(b * 2048 + row) * 1024 + col] =
          (__bf16)(o[j][r] * rowInv[wave][h8 * 8 + r]);
    }
  }
}

// ---------------------------------------------------------------------------
// Launch
// ---------------------------------------------------------------------------
extern "C" void kernel_launch(void* const* d_in, const int* in_sizes, int n_in,
                              void* d_out, int out_size, void* d_ws, size_t ws_size,
                              hipStream_t stream) {
  const float* x    = (const float*)d_in[0];   // [2,2048,1024]
  const float* Wqkv = (const float*)d_in[1];   // [1024,3072]
  const float* bqkv = (const float*)d_in[2];   // [3072]
  const float* Wout = (const float*)d_in[3];   // [1024,1024]
  const float* bout = (const float*)d_in[4];   // [1024]
  float* out = (float*)d_out;                  // [2,2048,1024]

  const size_t M = 4096;                       // B*N tokens
  __bf16* xb     = (__bf16*)d_ws;                    // [4096][1024]
  __bf16* wqkvT  = xb    + M * 1024;                 // [3072][1024] (W^T)
  __bf16* woutT  = wqkvT + (size_t)3072 * 1024;      // [1024][1024] (W^T)
  __bf16* qkv    = woutT + (size_t)1024 * 1024;      // [4096][3072] (Q,K thirds)
  __bf16* attn   = qkv   + M * 3072;                 // [4096][1024]
  __bf16* vT     = attn  + M * 1024;                 // [2][16][64][2048]

  cvt_f32_to_bf16<<<(M * 1024) / 1024, 256, 0, stream>>>(x, xb);
  cvt_transpose_bf16<<<dim3(3072 / 256, 1024), 256, 0, stream>>>(Wqkv, wqkvT, 3072, 1024);
  cvt_transpose_bf16<<<dim3(1024 / 256, 1024), 256, 0, stream>>>(Wout, woutT, 1024, 1024);

  // QKV projection: Q,K -> qkv; V -> vT (transposed)
  gemm_bf16_wmma<__bf16, true><<<dim3(3072 / 128, M / 128), 256, 0, stream>>>(
      xb, wqkvT, bqkv, qkv, vT, 1024, 1024, 1024, 3072);

  // Causal flash attention
  flash_attn_wmma<<<dim3(32, 16, 2), 128, 0, stream>>>(qkv, vT, attn);

  // Output projection -> f32 out
  gemm_bf16_wmma<float, false><<<dim3(1024 / 128, M / 128), 256, 0, stream>>>(
      attn, woutT, bout, out, nullptr, 1024, 1024, 1024, 1024);
}
